// GraphNet_27264452395684
// MI455X (gfx1250) — matrix operand order repaired and checked by reading, not compiled, not run on hardware
//
#include <hip/hip_runtime.h>
#include <hip/hip_bf16.h>

// ---------------------------------------------------------------------------
// GraphNet (8x EdgeConv + final FC) for MI455X / gfx1250.
// bf16 WMMA for the per-edge MLP GEMMs, folded BatchNorm, ordered-uint
// atomicMax for segment_max, pre-transposed (N-major) weights, and B
// (weight) fragments hoisted into registers across the M-tile loop.
// ---------------------------------------------------------------------------

typedef __attribute__((ext_vector_type(16))) __bf16 v16bf;
typedef __attribute__((ext_vector_type(8)))  __bf16 v8bf;
typedef __attribute__((ext_vector_type(8)))  float  v8f;

#define GN_L   8
#define GN_F   128
#define GN_C   256
#define GN_EPS 1e-5f
#define ET     64          // edges per workgroup (4 WMMA M-tiles)

// ---------------------------------------------------------------------------
// Prep: fold BN into scale/shift, convert weights to bf16 transposed N-major.
//   fc1 : [L][C][F] row-major float  ->  w1t : [L][F][C] bf16 (col-major of w1)
//   fc2 : [L][F][F] row-major float  ->  w2t : [L][F][F] bf16 (transposed)
// ---------------------------------------------------------------------------
__global__ __launch_bounds__(256) void gn_prep(
    const float* __restrict__ fc1, const float* __restrict__ fc2,
    const float* __restrict__ g1, const float* __restrict__ b1,
    const float* __restrict__ m1, const float* __restrict__ v1,
    const float* __restrict__ g2, const float* __restrict__ b2,
    const float* __restrict__ m2, const float* __restrict__ v2,
    __bf16* __restrict__ w1t, __bf16* __restrict__ w2t,
    float* __restrict__ s1, float* __restrict__ t1,
    float* __restrict__ s2, float* __restrict__ t2)
{
    const int i = blockIdx.x * 256 + threadIdx.x;

    // w1 transpose: i = (l*C + k)*F + n  ->  w1t[(l*F + n)*C + k]
    if (i < GN_L * GN_C * GN_F) {
        int n = i % GN_F;
        int k = (i / GN_F) % GN_C;
        int l = i / (GN_F * GN_C);
        w1t[((size_t)l * GN_F + n) * GN_C + k] = (__bf16)fc1[i];
    }
    // w2 transpose: i = (l*F + k)*F + n  ->  w2t[(l*F + n)*F + k]
    if (i < GN_L * GN_F * GN_F) {
        int n = i % GN_F;
        int k = (i / GN_F) % GN_F;
        int l = i / (GN_F * GN_F);
        w2t[((size_t)l * GN_F + n) * GN_F + k] = (__bf16)fc2[i];
    }
    if (i < GN_L * GN_C) {
        float s = g1[i] * rsqrtf(v1[i] + GN_EPS);
        s1[i] = s; t1[i] = b1[i] - m1[i] * s;
    }
    if (i < GN_L * GN_F) {
        float s = g2[i] * rsqrtf(v2[i] + GN_EPS);
        s2[i] = s; t2[i] = b2[i] - m2[i] * s;
    }
}

// ---------------------------------------------------------------------------
// Fused EdgeConv: gather + BN1/ReLU -> WMMA GEMM (Ex256 @ 256x128)
//              -> BN2/ReLU -> WMMA GEMM (Ex128 @ 128x128)
//              -> ordered-uint atomicMax segment aggregation.
// 256 threads = 8 waves; wave w owns output columns [16w, 16w+16).
// B-fragments are loaded once per wave and reused across all 4 M-tiles.
// ---------------------------------------------------------------------------
__global__ __launch_bounds__(256) void gn_edgeconv(
    const float* __restrict__ x,
    const int*   __restrict__ src, const int* __restrict__ dst,
    const __bf16* __restrict__ w1t,   // [F][C] N-major
    const __bf16* __restrict__ w2t,   // [F][F] N-major
    const float* __restrict__ s1, const float* __restrict__ t1,
    const float* __restrict__ s2, const float* __restrict__ t2,
    unsigned* __restrict__ agg, int E)
{
    __shared__ __bf16 eS[ET][GN_C + 8];  // post-BN1 edge features (bf16)
    __shared__ __bf16 hS[ET][GN_F + 8];  // post-BN2 hidden (bf16)
    __shared__ int    dS[ET];            // dst node per edge

    const int tid = threadIdx.x;

    // ---- Phase 1: gather + BN1 + ReLU -> LDS ------------------------------
    {
        const int el = tid >> 2, q = tid & 3;   // 4 threads per edge
        int e = blockIdx.x * ET + el;
        if (e >= E) e = E - 1;
        const int vi = dst[e], vj = src[e];
        if (q == 0) dS[el] = vi;
        const float4* xi4 = (const float4*)(x + (size_t)vi * GN_F);
        const float4* xj4 = (const float4*)(x + (size_t)vj * GN_F);
#pragma unroll
        for (int u = 0; u < 8; ++u) {
            const int c = q * 32 + u * 4;
            float4 a = xi4[c >> 2];
            float4 b = xj4[c >> 2];
            float av[4] = {a.x, a.y, a.z, a.w};
            float bv[4] = {b.x, b.y, b.z, b.w};
#pragma unroll
            for (int k = 0; k < 4; ++k) {
                const int cc = c + k;
                float e1 = fmaxf(av[k] * s1[cc] + t1[cc], 0.f);
                float e2 = fmaxf((bv[k] - av[k]) * s1[GN_F + cc] + t1[GN_F + cc], 0.f);
                eS[el][cc]        = (__bf16)e1;
                eS[el][GN_F + cc] = (__bf16)e2;
            }
        }
    }

    const int wave = tid >> 5, lane = tid & 31;
    const int half = lane >> 4, ln = lane & 15;
    const int n0   = wave * 16;
    const int ncol = n0 + ln;

    // Hoist GEMM1 B-fragments (invariant across M-tiles): 8 x v16bf.
    v16bf b1f[8];
#pragma unroll
    for (int ks = 0; ks < 8; ++ks)
        b1f[ks] = *(const v16bf*)(w1t + (size_t)ncol * GN_C + ks * 32 + 16 * half);

    const float s2n = s2[ncol];
    const float t2n = t2[ncol];

    __syncthreads();

    // ---- Phase 2: GEMM1 + BN2 + ReLU -> hS --------------------------------
#pragma unroll
    for (int mt = 0; mt < 4; ++mt) {
        v8f acc = {0.f, 0.f, 0.f, 0.f, 0.f, 0.f, 0.f, 0.f};
        const int row = mt * 16 + ln;
#pragma unroll
        for (int ks = 0; ks < 8; ++ks) {
            const int kb = ks * 32;
            // A-fragment: two contiguous 128-bit LDS loads per lane
            v8bf alo = *(const v8bf*)&eS[row][kb + 16 * half];
            v8bf ahi = *(const v8bf*)&eS[row][kb + 16 * half + 8];
            v16bf a = __builtin_shufflevector(alo, ahi,
                0, 1, 2, 3, 4, 5, 6, 7, 8, 9, 10, 11, 12, 13, 14, 15);
            acc = __builtin_amdgcn_wmma_f32_16x16x32_bf16(
                false, a, false, b1f[ks], (short)0, acc, false, false);
        }
#pragma unroll
        for (int r = 0; r < 8; ++r) {
            const int m = mt * 16 + r + 8 * half;
            hS[m][ncol] = (__bf16)fmaxf(acc[r] * s2n + t2n, 0.f);
        }
    }

    // Hoist GEMM2 B-fragments: 4 x v16bf.
    v16bf b2f[4];
#pragma unroll
    for (int ks = 0; ks < 4; ++ks)
        b2f[ks] = *(const v16bf*)(w2t + (size_t)ncol * GN_F + ks * 32 + 16 * half);

    __syncthreads();

    // ---- Phase 3: GEMM2 + atomic segment-max ------------------------------
#pragma unroll
    for (int mt = 0; mt < 4; ++mt) {
        v8f acc = {0.f, 0.f, 0.f, 0.f, 0.f, 0.f, 0.f, 0.f};
        const int row = mt * 16 + ln;
#pragma unroll
        for (int ks = 0; ks < 4; ++ks) {
            const int kb = ks * 32;
            v8bf alo = *(const v8bf*)&hS[row][kb + 16 * half];
            v8bf ahi = *(const v8bf*)&hS[row][kb + 16 * half + 8];
            v16bf a = __builtin_shufflevector(alo, ahi,
                0, 1, 2, 3, 4, 5, 6, 7, 8, 9, 10, 11, 12, 13, 14, 15);
            acc = __builtin_amdgcn_wmma_f32_16x16x32_bf16(
                false, a, false, b2f[ks], (short)0, acc, false, false);
        }
#pragma unroll
        for (int r = 0; r < 8; ++r) {
            const int m  = mt * 16 + r + 8 * half;
            const int eg = blockIdx.x * ET + m;
            if (eg < E) {
                unsigned bits = __float_as_uint(acc[r]);
                // order-preserving float->uint; 0 is reserved (below key(-inf))
                unsigned key = bits ^ ((unsigned)((int)bits >> 31) | 0x80000000u);
                atomicMax(&agg[(size_t)dS[m] * GN_F + ncol], key);
            }
        }
    }
}

// ---------------------------------------------------------------------------
// Fixup: decode ordered-uint max, empty segments -> 0, optional residual add.
// ---------------------------------------------------------------------------
__global__ __launch_bounds__(256) void gn_fixup(
    const unsigned* __restrict__ agg, const float* __restrict__ res,
    float* __restrict__ out, int total)
{
    const int i = blockIdx.x * 256 + threadIdx.x;
    if (i >= total) return;
    const unsigned key = agg[i];
    float v = 0.f;
    if (key != 0u) {
        unsigned bits = (key & 0x80000000u) ? (key ^ 0x80000000u) : ~key;
        v = __uint_as_float(bits);
    }
    if (res) v += res[i];
    out[i] = v;
}

// ---------------------------------------------------------------------------
// Final FC: [N,128] @ [128,2] + bias.
// ---------------------------------------------------------------------------
__global__ __launch_bounds__(256) void gn_final(
    const float* __restrict__ h, const float* __restrict__ fcw,
    const float* __restrict__ fcb, float* __restrict__ out, int n)
{
    const int i = blockIdx.x * 256 + threadIdx.x;
    if (i >= n) return;
    float a0 = fcb[0], a1 = fcb[1];
    const float* r = h + (size_t)i * GN_F;
#pragma unroll 4
    for (int c = 0; c < GN_F; ++c) {
        float xv = r[c];
        a0 += xv * fcw[c * 2 + 0];
        a1 += xv * fcw[c * 2 + 1];
    }
    out[i * 2 + 0] = a0;
    out[i * 2 + 1] = a1;
}

// ---------------------------------------------------------------------------
extern "C" void kernel_launch(void* const* d_in, const int* in_sizes, int n_in,
                              void* d_out, int out_size, void* d_ws, size_t ws_size,
                              hipStream_t stream) {
    (void)n_in; (void)out_size; (void)ws_size;

    const float* x    = (const float*)d_in[0];
    const int*   sei  = (const int*)d_in[1];
    const int*   tei  = (const int*)d_in[2];
    const float* g1   = (const float*)d_in[3];
    const float* b1   = (const float*)d_in[4];
    const float* m1   = (const float*)d_in[5];
    const float* v1   = (const float*)d_in[6];
    const float* fc1  = (const float*)d_in[7];
    const float* g2   = (const float*)d_in[8];
    const float* b2   = (const float*)d_in[9];
    const float* m2   = (const float*)d_in[10];
    const float* v2   = (const float*)d_in[11];
    const float* fc2  = (const float*)d_in[12];
    const float* fcw  = (const float*)d_in[13];
    const float* fcb  = (const float*)d_in[14];

    const int E  = in_sizes[1] / 2;       // 800000
    const int Nn = in_sizes[0] / GN_F;    // 50000
    const size_t NF = (size_t)Nn * GN_F;

    // Workspace carve-out (256B-aligned slabs).
    char* ws = (char*)d_ws;
    size_t off = 0;
    auto take = [&](size_t bytes) -> char* {
        char* p = ws + off;
        off = (off + bytes + 255) & ~(size_t)255;
        return p;
    };
    __bf16*   w1t  = (__bf16*)take((size_t)GN_L * GN_C * GN_F * sizeof(__bf16));
    __bf16*   w2t  = (__bf16*)take((size_t)GN_L * GN_F * GN_F * sizeof(__bf16));
    float*    s1   = (float*)take((size_t)GN_L * GN_C * sizeof(float));
    float*    t1   = (float*)take((size_t)GN_L * GN_C * sizeof(float));
    float*    s2   = (float*)take((size_t)GN_L * GN_F * sizeof(float));
    float*    t2   = (float*)take((size_t)GN_L * GN_F * sizeof(float));
    unsigned* agg  = (unsigned*)take(NF * sizeof(unsigned));
    float*    bufA = (float*)take(NF * sizeof(float));
    float*    bufB = (float*)take(NF * sizeof(float));
    float*    bufC = (float*)take(NF * sizeof(float));

    gn_prep<<<(GN_L * GN_C * GN_F + 255) / 256, 256, 0, stream>>>(
        fc1, fc2, g1, b1, m1, v1, g2, b2, m2, v2, w1t, w2t, s1, t1, s2, t2);

    // Layer schedule (residual every block after the first):
    //   L0: x   ->A          L1: A ->B
    //   L2: B   ->A          L3: A(+B)->C
    //   L4: C   ->A          L5: A(+C)->B
    //   L6: B   ->A          L7: A(+B)->C
    const float* ins[GN_L]  = {x, bufA, bufB, bufA, bufC, bufA, bufB, bufA};
    float*       outs[GN_L] = {bufA, bufB, bufA, bufC, bufA, bufB, bufA, bufC};
    const float* ress[GN_L] = {nullptr, nullptr, nullptr, bufB,
                               nullptr, bufC, nullptr, bufB};

    const int egrid = (E + ET - 1) / ET;
    const int fgrid = (int)((NF + 255) / 256);

    for (int l = 0; l < GN_L; ++l) {
        const int* ei  = (l & 1) ? tei : sei;
        const int* sr  = ei;          // row 0 = src
        const int* dsn = ei + E;      // row 1 = dst
        hipMemsetAsync(agg, 0, NF * sizeof(unsigned), stream);
        gn_edgeconv<<<egrid, 256, 0, stream>>>(
            ins[l], sr, dsn,
            w1t + (size_t)l * GN_C * GN_F, w2t + (size_t)l * GN_F * GN_F,
            s1 + (size_t)l * GN_C, t1 + (size_t)l * GN_C,
            s2 + (size_t)l * GN_F, t2 + (size_t)l * GN_F,
            agg, E);
        gn_fixup<<<fgrid, 256, 0, stream>>>(agg, ress[l], outs[l], (int)NF);
    }

    gn_final<<<(Nn + 255) / 256, 256, 0, stream>>>(
        bufC, fcw, fcb, (float*)d_out, Nn);
}
